// ShiftWiseConv_15272903705343
// MI455X (gfx1250) — matrix operand order repaired and checked by reading, not compiled, not run on hardware
//
#include <hip/hip_runtime.h>

// ---------------------------------------------------------------------------
// ShiftWiseConv on MI455X (gfx1250, wave32, WMMA, TDM).
// Pipeline:
//   cvt weights -> bf16 (once)
//   expand GEMM  (bf16 WMMA, TDM-staged B panel, double-buffered A tile,
//                 LDS-transposed coalesced bf16 store)
//   roll + dw3x3 + sum  (LDS row staging)
//   mix GEMM + BN + SiLU (same GEMM core, coalesced f32 store)
// Intermediates in d_ws as bf16; x / x_exp re-reads stay in the 192MB L2.
// ---------------------------------------------------------------------------

#define B_IMG   16
#define C1      256
#define C2      256
#define NK      5
#define EXP_CH  (C2 * NK)   // 1280
#define HW      4096        // 64*64
#define W_IMG   64
#define BN_EPS  1e-5f

typedef __attribute__((ext_vector_type(16))) __bf16        bf16x16;
typedef __attribute__((ext_vector_type(8)))  float         f32x8;
typedef __attribute__((ext_vector_type(4)))  unsigned int  tdm_v4u;
typedef __attribute__((ext_vector_type(8)))  int           tdm_v8i;
typedef __attribute__((ext_vector_type(4)))  int           tdm_v4i;

union Frag16 { unsigned int u[8]; bf16x16 v; };

#if defined(__gfx1250__) && __has_builtin(__builtin_amdgcn_tensor_load_to_lds)
#define USE_TDM 1
#else
#define USE_TDM 0
#endif

__device__ __forceinline__ unsigned short f2bf(float f) {
    unsigned int u = __float_as_uint(f);
    u += 0x7FFFu + ((u >> 16) & 1u);          // round-to-nearest-even
    return (unsigned short)(u >> 16);
}
__device__ __forceinline__ float bf2f(unsigned short h) {
    return __uint_as_float(((unsigned int)h) << 16);
}

#if USE_TDM
// 2-D bf16 tile load, Global -> LDS, via Tensor Data Mover (6-arg builtin).
// D# per CDNA5 ISA: group0 {count=1, lds_addr, global_addr,
// type=2}; group1 {data_size=2B, pad 1 dword every 128 dwords, dims/strides}.
// Row image in LDS: tile_k bf16 + 1 dword pad (tile_k=256 -> stride 258 u16).
__device__ __forceinline__ void tdm_load_bf16_2d(unsigned lds_off_bytes,
                                                 const unsigned short* gptr,
                                                 unsigned tile_k,  // dim0 elems
                                                 unsigned tile_n,  // dim1 rows
                                                 unsigned long long stride_el,
                                                 unsigned pad_interval_code,
                                                 unsigned pad_amount_code) {
    unsigned long long ga = (unsigned long long)(const void*)gptr;
    tdm_v4u g0;
    g0.x = 1u;                                           // count=1 (valid D#)
    g0.y = lds_off_bytes;                                // lds_addr
    g0.z = (unsigned)(ga & 0xFFFFFFFFu);                 // global_addr[31:0]
    g0.w = (unsigned)((ga >> 32) & 0x01FFFFFFu)          // global_addr[56:32]
         | (2u << 30);                                   // type = 2 ("image")
    tdm_v8i g1;
    g1[0] = (int)((1u << 16)                             // data_size = 2 bytes
                | (1u << 20)                             // pad_enable
                | (pad_interval_code << 22)
                | (pad_amount_code << 25));
    g1[1] = (int)((tile_k & 0xFFFFu) << 16);             // tensor_dim0 lo
    g1[2] = (int)((tile_k >> 16) | ((tile_n & 0xFFFFu) << 16)); // td0 hi|td1 lo
    g1[3] = (int)((tile_n >> 16) | (tile_k << 16));      // td1 hi | tile_dim0
    g1[4] = (int)tile_n;                                 // tile_dim1, dim2=0
    g1[5] = (int)(stride_el & 0xFFFFFFFFu);              // dim0_stride lo
    g1[6] = (int)((stride_el >> 32) & 0xFFFFu);          // dim0_stride hi
    g1[7] = 0;
    tdm_v4i gz4 = {0, 0, 0, 0};
    tdm_v8i gz8 = {0, 0, 0, 0, 0, 0, 0, 0};
    __builtin_amdgcn_tensor_load_to_lds(g0, g1, gz4, gz4, gz8, 0);
}
#endif

// ---------------------------------------------------------------------------
// Kernel: f32 -> bf16 array convert (weights are tiny; done once).
// ---------------------------------------------------------------------------
__global__ void cvt_bf16_kernel(const float* __restrict__ src,
                                unsigned short* __restrict__ dst, int n) {
    int i = blockIdx.x * blockDim.x + threadIdx.x;
    if (i < n) dst[i] = f2bf(src[i]);
}

// ---------------------------------------------------------------------------
// Kernel: fold 3 depthwise paths -> w_eff[k][c][9]
// ---------------------------------------------------------------------------
__global__ void weff_kernel(const float* __restrict__ w_path,
                            float* __restrict__ weff) {
    int idx = blockIdx.x * blockDim.x + threadIdx.x;
    if (idx >= NK * C2 * 9) return;
    int t = idx % 9;
    int kc = idx / 9;
    int c = kc % C2;
    int k = kc / C2;
    float s = 0.f;
#pragma unroll
    for (int p = 0; p < 3; ++p)
        s += w_path[((size_t)(p * NK + k) * C2 + c) * 9 + t];
    weff[idx] = s;
}

// ---------------------------------------------------------------------------
// Shared GEMM tiling: block = 64(M) x 128(N), 8 waves (wm 0..3, wn 0..1),
// each wave: one 16-row M tile x four 16-col N tiles; K step 32 (bf16 WMMA).
// B panel (128 x 256 bf16, rows padded to 258 u16) resident in LDS (TDM).
// A tile double-buffered: next tile's global loads issue before the WMMAs,
// LDS writes after the fragment reads -> one barrier per K iteration.
// ---------------------------------------------------------------------------
#define BP_STRIDE 258          // u16 per B-panel row (256 data + 1 dword pad)

// ---------------- Kernel: expand 1x1 conv (x @ w_expand^T) -----------------
__global__ void __launch_bounds__(256)
expand_gemm_kernel(const float* __restrict__ x,
                   const unsigned short* __restrict__ wexp_bf,
                   unsigned short* __restrict__ xexp) {
    __shared__ union {
        struct {
            unsigned short A[2][64][34];       // [buf][m][k] double buffer
            unsigned short B[128][BP_STRIDE];  // [n][k] full-K panel
        } in;
        unsigned short C[128][72];             // [co][m] store staging
    } sh;

    const int tid  = threadIdx.x;
    const int lane = tid & 31;
    const int wave = tid >> 5;
    const int wm   = wave >> 1;            // 0..3
    const int wn   = wave & 1;             // 0..1
    const int m0   = blockIdx.x * 64;      // 64 | 4096: tile within one image
    const int n0   = blockIdx.y * 128;
    const int bi   = m0 >> 12;             // batch image (constant per block)
    const int p0   = m0 & 4095;

    // per-thread A staging indices (64m x 32k tile as 2 x float4 per thread)
    const int kkA0 = tid >> 4;             // 0..15
    const int kkA1 = kkA0 + 16;            // 16..31
    const int mqA  = (tid & 15) << 2;      // 0..60, 4 consecutive pixels

    // ---- load full B panel (128 x 256 bf16) once ----
#if USE_TDM
    if (wave == 0) {
        tdm_load_bf16_2d((unsigned)(unsigned long long)&sh.in.B[0][0],
                         wexp_bf + (size_t)n0 * C1,
                         /*tile_k=*/C1, /*tile_n=*/128,
                         /*stride=*/C1, /*pad=128dw interval*/6u, /*1dw*/0u);
        __builtin_amdgcn_s_wait_tensorcnt(0);
    }
#else
    for (int i = tid; i < 128 * C1; i += 256) {
        int nn = i >> 8, kk = i & 255;
        sh.in.B[nn][kk] = wexp_bf[(size_t)(n0 + nn) * C1 + kk];
    }
#endif

    // ---- prologue: stage A tile for k0 = 0 into buffer 0 ----
    {
        float4 v0 = *(const float4*)&x[((size_t)bi * C1 + kkA0) * HW + p0 + mqA];
        float4 v1 = *(const float4*)&x[((size_t)bi * C1 + kkA1) * HW + p0 + mqA];
        sh.in.A[0][mqA + 0][kkA0] = f2bf(v0.x);
        sh.in.A[0][mqA + 1][kkA0] = f2bf(v0.y);
        sh.in.A[0][mqA + 2][kkA0] = f2bf(v0.z);
        sh.in.A[0][mqA + 3][kkA0] = f2bf(v0.w);
        sh.in.A[0][mqA + 0][kkA1] = f2bf(v1.x);
        sh.in.A[0][mqA + 1][kkA1] = f2bf(v1.y);
        sh.in.A[0][mqA + 2][kkA1] = f2bf(v1.z);
        sh.in.A[0][mqA + 3][kkA1] = f2bf(v1.w);
    }
    __syncthreads();

    f32x8 acc[4];
#pragma unroll
    for (int t = 0; t < 4; ++t)
        acc[t] = (f32x8){0.f, 0.f, 0.f, 0.f, 0.f, 0.f, 0.f, 0.f};

    for (int k0 = 0; k0 < C1; k0 += 32) {
        const int buf = (k0 >> 5) & 1;
        const bool has_next = (k0 + 32) < C1;

        // issue next A tile's global loads early (overlap with WMMAs)
        float4 v0, v1;
        if (has_next) {
            v0 = *(const float4*)
                 &x[((size_t)bi * C1 + (k0 + 32 + kkA0)) * HW + p0 + mqA];
            v1 = *(const float4*)
                 &x[((size_t)bi * C1 + (k0 + 32 + kkA1)) * HW + p0 + mqA];
        }

        // A fragment (16x32 bf16, ISA 7.12.2)
        Frag16 fa;
        {
            const int row = lane & 15;
            const int kb  = (lane >> 4) << 3;
#pragma unroll
            for (int j = 0; j < 8; ++j) {
                int k = kb + ((j & 3) << 1) + ((j >> 2) << 4);
                fa.u[j] = *(const unsigned int*)&sh.in.A[buf][wm * 16 + row][k];
            }
        }
        const int col = lane & 15;
        const int kbB = (lane >> 4) << 3;
#pragma unroll
        for (int t = 0; t < 4; ++t) {
            Frag16 fb;
#pragma unroll
            for (int j = 0; j < 8; ++j) {
                int k = k0 + kbB + ((j & 3) << 1) + ((j >> 2) << 4);
                fb.u[j] = *(const unsigned int*)
                          &sh.in.B[wn * 64 + t * 16 + col][k];
            }
            acc[t] = __builtin_amdgcn_wmma_f32_16x16x32_bf16(
                false, fa.v, false, fb.v, (short)0, acc[t], false, false);
        }

        // write next tile into the alternate buffer (after fragment reads)
        if (has_next) {
            const int nb = buf ^ 1;
            sh.in.A[nb][mqA + 0][kkA0] = f2bf(v0.x);
            sh.in.A[nb][mqA + 1][kkA0] = f2bf(v0.y);
            sh.in.A[nb][mqA + 2][kkA0] = f2bf(v0.z);
            sh.in.A[nb][mqA + 3][kkA0] = f2bf(v0.w);
            sh.in.A[nb][mqA + 0][kkA1] = f2bf(v1.x);
            sh.in.A[nb][mqA + 1][kkA1] = f2bf(v1.y);
            sh.in.A[nb][mqA + 2][kkA1] = f2bf(v1.z);
            sh.in.A[nb][mqA + 3][kkA1] = f2bf(v1.w);
        }
        __syncthreads();
    }

    // ---- stage C into LDS [co][m], then coalesced 128b stores ----
    {
        const int mrow = (lane >> 4) << 3;
        const int ncol = lane & 15;
#pragma unroll
        for (int t = 0; t < 4; ++t)
#pragma unroll
            for (int r = 0; r < 8; ++r)
                sh.C[wn * 64 + t * 16 + ncol][wm * 16 + mrow + r] =
                    f2bf(acc[t][r]);
    }
    __syncthreads();
    {
        const int co_l = tid >> 1;          // 0..127
        const int half = tid & 1;           // 32 pixels each
        unsigned short* dst = xexp +
            ((size_t)bi * EXP_CH + (n0 + co_l)) * HW + p0 + half * 32;
        const uint4* srcv = (const uint4*)&sh.C[co_l][half * 32];
#pragma unroll
        for (int i = 0; i < 4; ++i)
            ((uint4*)dst)[i] = srcv[i];
    }
}

// ---------------------------------------------------------------------------
// Kernel: roll + zero-padded dw3x3 + sum over k.  Shift for group k = 3k-4.
// Block = 4 rows (256 px) of one (b, c) plane.  Rolled source rows staged in
// LDS with coalesced row copies; taps read LDS.
// ---------------------------------------------------------------------------
__global__ void __launch_bounds__(256)
dwconv_kernel(const unsigned short* __restrict__ xexp,
              const float* __restrict__ weff,
              unsigned short* __restrict__ merged) {
    __shared__ float          wl[NK][9];
    __shared__ unsigned short lrow[NK][6][W_IMG];   // 6 rows = 4 out + halo

    const int tid = threadIdx.x;
    const int c   = blockIdx.x & (C2 - 1);
    const int b   = blockIdx.x >> 8;
    const int y0  = blockIdx.y * 4;

    if (tid < NK * 9)
        wl[tid / 9][tid % 9] = weff[((tid / 9) * C2 + c) * 9 + (tid % 9)];

    // stage rolled rows: lrow[k][j][sx] = src_k[((y0-1+j) - s_k) & 63][sx]
    for (int i = tid; i < NK * 6 * W_IMG; i += 256) {
        int k  = i / (6 * W_IMG);
        int r  = i % (6 * W_IMG);
        int j  = r >> 6;
        int sx = r & 63;
        int s  = 3 * k - 4;
        int sy = (y0 - 1 + j - s) & 63;
        lrow[k][j][sx] =
            xexp[((size_t)b * EXP_CH + c * NK + k) * HW + sy * W_IMG + sx];
    }
    __syncthreads();

    const int ry = tid >> 6;        // 0..3 row within block
    const int xq = tid & 63;
    const int y  = y0 + ry;

    float acc = 0.f;
#pragma unroll
    for (int k = 0; k < NK; ++k) {
        const int s = 3 * k - 4;
#pragma unroll
        for (int dy = 0; dy < 3; ++dy) {
            int yy = y + dy - 1;
            if ((unsigned)yy >= (unsigned)W_IMG) continue;   // zero pad
            int j = ry + dy;
#pragma unroll
            for (int dx = 0; dx < 3; ++dx) {
                int xx = xq + dx - 1;
                if ((unsigned)xx >= (unsigned)W_IMG) continue;
                int sx = (xx - s) & 63;                       // roll
                acc += wl[k][dy * 3 + dx] * bf2f(lrow[k][j][sx]);
            }
        }
    }
    merged[((size_t)b * C2 + c) * HW + y * W_IMG + xq] = f2bf(acc);
}

// ---------------- Kernel: mix 1x1 conv + BN + SiLU, f32 out ----------------
__global__ void __launch_bounds__(256)
mix_gemm_kernel(const unsigned short* __restrict__ merged,
                const unsigned short* __restrict__ wmix_bf,
                const float* __restrict__ bn_gamma,
                const float* __restrict__ bn_beta,
                const float* __restrict__ bn_mean,
                const float* __restrict__ bn_var,
                float* __restrict__ out) {
    __shared__ union {
        struct {
            unsigned short A[2][64][34];
            unsigned short B[128][BP_STRIDE];
        } in;
        float C[128][68];                  // [co][m] f32 staging
    } sh;

    const int tid  = threadIdx.x;
    const int lane = tid & 31;
    const int wave = tid >> 5;
    const int wm   = wave >> 1;
    const int wn   = wave & 1;
    const int m0   = blockIdx.x * 64;
    const int n0   = blockIdx.y * 128;
    const int bi   = m0 >> 12;
    const int p0   = m0 & 4095;

    // per-thread A staging: 64m x 32k bf16 tile as 1 x uint4 per thread
    const int kkA = tid >> 3;              // 0..31
    const int mqA = (tid & 7) << 3;        // 0..56, 8 consecutive pixels

#if USE_TDM
    if (wave == 0) {
        tdm_load_bf16_2d((unsigned)(unsigned long long)&sh.in.B[0][0],
                         wmix_bf + (size_t)n0 * C2,
                         C2, 128, C2, 6u, 0u);
        __builtin_amdgcn_s_wait_tensorcnt(0);
    }
#else
    for (int i = tid; i < 128 * C2; i += 256) {
        int nn = i >> 8, kk = i & 255;
        sh.in.B[nn][kk] = wmix_bf[(size_t)(n0 + nn) * C2 + kk];
    }
#endif

    // prologue: stage A tile for k0 = 0
    {
        uint4 v = *(const uint4*)&merged[((size_t)bi * C2 + kkA) * HW + p0 + mqA];
        const unsigned int uu[4] = {v.x, v.y, v.z, v.w};
#pragma unroll
        for (int j = 0; j < 4; ++j) {
            sh.in.A[0][mqA + 2 * j + 0][kkA] = (unsigned short)(uu[j] & 0xFFFFu);
            sh.in.A[0][mqA + 2 * j + 1][kkA] = (unsigned short)(uu[j] >> 16);
        }
    }
    __syncthreads();

    f32x8 acc[4];
#pragma unroll
    for (int t = 0; t < 4; ++t)
        acc[t] = (f32x8){0.f, 0.f, 0.f, 0.f, 0.f, 0.f, 0.f, 0.f};

    for (int k0 = 0; k0 < C2; k0 += 32) {
        const int buf = (k0 >> 5) & 1;
        const bool has_next = (k0 + 32) < C2;

        uint4 v;
        if (has_next)
            v = *(const uint4*)
                &merged[((size_t)bi * C2 + (k0 + 32 + kkA)) * HW + p0 + mqA];

        Frag16 fa;
        {
            const int row = lane & 15;
            const int kb  = (lane >> 4) << 3;
#pragma unroll
            for (int j = 0; j < 8; ++j) {
                int k = kb + ((j & 3) << 1) + ((j >> 2) << 4);
                fa.u[j] = *(const unsigned int*)&sh.in.A[buf][wm * 16 + row][k];
            }
        }
        const int col = lane & 15;
        const int kbB = (lane >> 4) << 3;
#pragma unroll
        for (int t = 0; t < 4; ++t) {
            Frag16 fb;
#pragma unroll
            for (int j = 0; j < 8; ++j) {
                int k = k0 + kbB + ((j & 3) << 1) + ((j >> 2) << 4);
                fb.u[j] = *(const unsigned int*)
                          &sh.in.B[wn * 64 + t * 16 + col][k];
            }
            acc[t] = __builtin_amdgcn_wmma_f32_16x16x32_bf16(
                false, fa.v, false, fb.v, (short)0, acc[t], false, false);
        }

        if (has_next) {
            const int nb = buf ^ 1;
            const unsigned int uu[4] = {v.x, v.y, v.z, v.w};
#pragma unroll
            for (int j = 0; j < 4; ++j) {
                sh.in.A[nb][mqA + 2 * j + 0][kkA] =
                    (unsigned short)(uu[j] & 0xFFFFu);
                sh.in.A[nb][mqA + 2 * j + 1][kkA] =
                    (unsigned short)(uu[j] >> 16);
            }
        }
        __syncthreads();
    }

    // stage raw acc to LDS [co][m]
    {
        const int mrow = (lane >> 4) << 3;
        const int ncol = lane & 15;
#pragma unroll
        for (int t = 0; t < 4; ++t)
#pragma unroll
            for (int r = 0; r < 8; ++r)
                sh.C[wn * 64 + t * 16 + ncol][wm * 16 + mrow + r] = acc[t][r];
    }
    __syncthreads();

    // BN + SiLU per channel row, coalesced float4 stores
    {
        const int co_l = tid >> 1;
        const int half = tid & 1;
        const int co   = n0 + co_l;
        const float sc = bn_gamma[co] * rsqrtf(bn_var[co] + BN_EPS);
        const float mu = bn_mean[co];
        const float be = bn_beta[co];
        float* dst = out + ((size_t)bi * C2 + co) * HW + p0 + half * 32;
        const float4* srcv = (const float4*)&sh.C[co_l][half * 32];
#pragma unroll
        for (int g = 0; g < 8; ++g) {
            float4 v = srcv[g];
            float4 o;
            o.x = (v.x - mu) * sc + be;  o.x = o.x / (1.f + __expf(-o.x));
            o.y = (v.y - mu) * sc + be;  o.y = o.y / (1.f + __expf(-o.y));
            o.z = (v.z - mu) * sc + be;  o.z = o.z / (1.f + __expf(-o.z));
            o.w = (v.w - mu) * sc + be;  o.w = o.w / (1.f + __expf(-o.w));
            ((float4*)dst)[g] = o;
        }
    }
}

// ---------------------------------------------------------------------------
// Host launcher
// ---------------------------------------------------------------------------
extern "C" void kernel_launch(void* const* d_in, const int* in_sizes, int n_in,
                              void* d_out, int out_size, void* d_ws, size_t ws_size,
                              hipStream_t stream) {
    const float* x        = (const float*)d_in[0];
    const float* w_expand = (const float*)d_in[1];
    const float* w_path   = (const float*)d_in[2];
    const float* w_mix    = (const float*)d_in[3];
    const float* bn_gamma = (const float*)d_in[4];
    const float* bn_beta  = (const float*)d_in[5];
    const float* bn_mean  = (const float*)d_in[6];
    const float* bn_var   = (const float*)d_in[7];
    float* out = (float*)d_out;

    // workspace layout (bytes):
    //   [0,       46080)   w_eff f32
    //   [64K,     +640K)   w_expand bf16 (1280 x 256)
    //   [704K,    +128K)   w_mix    bf16 (256 x 256)
    //   [1M,      +160M)   xexp  bf16 (16 x 1280 x 4096)
    //   [1M+160M, +32M )   merged bf16 (16 x 256 x 4096)
    char* ws = (char*)d_ws;
    float* weff = (float*)ws;
    unsigned short* wexp_bf = (unsigned short*)(ws + (size_t)64 * 1024);
    unsigned short* wmix_bf = (unsigned short*)(ws + (size_t)704 * 1024);
    unsigned short* xexp    = (unsigned short*)(ws + ((size_t)1 << 20));
    unsigned short* merged  = (unsigned short*)
        (ws + ((size_t)1 << 20) + (size_t)B_IMG * EXP_CH * HW * 2);

    cvt_bf16_kernel<<<(EXP_CH * C1 + 255) / 256, 256, 0, stream>>>(
        w_expand, wexp_bf, EXP_CH * C1);
    cvt_bf16_kernel<<<(C2 * C2 + 255) / 256, 256, 0, stream>>>(
        w_mix, wmix_bf, C2 * C2);
    weff_kernel<<<(NK * C2 * 9 + 255) / 256, 256, 0, stream>>>(w_path, weff);

    expand_gemm_kernel<<<dim3(B_IMG * HW / 64, EXP_CH / 128), 256, 0, stream>>>(
        x, wexp_bf, xexp);

    dwconv_kernel<<<dim3(B_IMG * C2, HW / 256), 256, 0, stream>>>(
        xexp, weff, merged);

    mix_gemm_kernel<<<dim3(B_IMG * HW / 64, C2 / 128), 256, 0, stream>>>(
        merged, wmix_bf, bn_gamma, bn_beta, bn_mean, bn_var, out);
}
